// ContrastiveLoss_69174743269559
// MI455X (gfx1250) — matrix operand order repaired and checked by reading, not compiled, not run on hardware
//
#include <hip/hip_runtime.h>

typedef _Float16 v16h __attribute__((ext_vector_type(16)));
typedef _Float16 v8h  __attribute__((ext_vector_type(8)));
typedef float    v8f  __attribute__((ext_vector_type(8)));

#define N_ROWS 8192              // B*T*H*W = 8*16*8*8
#define NF     128               // feature dim
#define N_TILES (N_ROWS / 16)    // 512
#define INV_T  10.0f             // 1/TEMPERATURE; also fixed logsumexp shift (max sim)
// exp(s - 10) = exp2(c * 10*log2e - 10*log2e), c = cosine sim
#define EXP2_SCALE 14.42695040888963f
#define EXP2_BIAS  (-14.42695040888963f)

__device__ __forceinline__ float fast_exp2(float x) {
#if __has_builtin(__builtin_amdgcn_exp2f)
  return __builtin_amdgcn_exp2f(x);   // v_exp_f32 directly
#else
  return exp2f(x);
#endif
}

// Load the A fragment (16x32 f16) for a row-tile, K-block kb.
// Lane<16 holds K 0..15; lane>=16 holds K 16..31 (contiguous).
__device__ __forceinline__ v16h load_afrag(const _Float16* __restrict__ arow,
                                           int kb, int hi) {
  v8h lo = *(const v8h*)(arow + kb * 32 + hi * 16);
  v8h hp = *(const v8h*)(arow + kb * 32 + hi * 16 + 8);
  return __builtin_shufflevector(lo, hp, 0, 1, 2, 3, 4, 5, 6, 7,
                                 8, 9, 10, 11, 12, 13, 14, 15);
}

// ---------------------------------------------------------------------------
// Kernel 1: row L2-normalization  x (f32, (8,128,16,8,8)) -> xn (f16, [8192,128])
// feats[n, f] = x[b*131072 + f*1024 + m],  n = b*1024 + m
// One wave per row; lane covers f = lane + {0,32,64,96}. Also zeroes the
// global accumulator (stream-ordered ahead of the WMMA kernels).
// ---------------------------------------------------------------------------
__global__ __launch_bounds__(256) void cl_normalize_kernel(
    const float* __restrict__ x, _Float16* __restrict__ xn, float* __restrict__ acc) {
  if (blockIdx.x == 0 && threadIdx.x == 0) *acc = 0.0f;

  const int wave = threadIdx.x >> 5;
  const int lane = threadIdx.x & 31;
  const int n = blockIdx.x * 8 + wave;        // 1024 blocks * 8 waves = 8192 rows
  const int b = n >> 10;
  const int m = n & 1023;

  const float* src = x + (size_t)b * 131072 + m;
  float v0 = src[(size_t)(lane +  0) * 1024];
  float v1 = src[(size_t)(lane + 32) * 1024];
  float v2 = src[(size_t)(lane + 64) * 1024];
  float v3 = src[(size_t)(lane + 96) * 1024];

  float ss = v0 * v0 + v1 * v1 + v2 * v2 + v3 * v3;
#pragma unroll
  for (int off = 16; off > 0; off >>= 1) ss += __shfl_xor(ss, off, 32);

  const float nrm = fmaxf(sqrtf(ss), 1e-12f);   // matches F.normalize clamp_min
  const float inv = 1.0f / nrm;

  _Float16* dst = xn + (size_t)n * NF;
  dst[lane +  0] = (_Float16)(v0 * inv);
  dst[lane + 32] = (_Float16)(v1 * inv);
  dst[lane + 64] = (_Float16)(v2 * inv);
  dst[lane + 96] = (_Float16)(v3 * inv);
}

// ---------------------------------------------------------------------------
// Unmasked sweep of row-tile `it` over column-tiles jt = it+wave, +8, ...
// Accumulates exp2(c*S + B) for ALL 256 elements of every tile (diagonal
// correction is applied exactly by cl_diagtile_kernel). Branch-free body;
// 4 independent partial sums break the add dependency chain so the adds
// dual-issue with the v_exp_f32 stream under the WMMAs.
// ---------------------------------------------------------------------------
__device__ __forceinline__ void cl_sweep_rowtile(
    const _Float16* __restrict__ xn, int it, int wave, int sub, int hi,
    float acc4[4]) {
  const _Float16* arow = xn + (size_t)(it * 16 + sub) * NF;
  v16h afrag[4];
#pragma unroll
  for (int kb = 0; kb < 4; ++kb) afrag[kb] = load_afrag(arow, kb, hi);

  // B fragment (32x16 f16): lane%16 = N (column = row jt*16+sub of X),
  // lanes 0-15 hold K 0..15, lanes 16-31 hold K 16..31 (contiguous 32B).
  const _Float16* brow = xn + (size_t)((it + wave) * 16 + sub) * NF + hi * 16;

  for (int jt = it + wave; jt < N_TILES; jt += 8, brow += 8 * 16 * NF) {
    v16h bfrag[4];
#pragma unroll
    for (int kb = 0; kb < 4; ++kb)            // clause of 8 b128 loads
      bfrag[kb] = *(const v16h*)(brow + kb * 32);

    v8f c = {};
#pragma unroll
    for (int kb = 0; kb < 4; ++kb)
      c = __builtin_amdgcn_wmma_f32_16x16x32_f16(
              false, afrag[kb], false, bfrag[kb], (short)0, c, false, false);

#pragma unroll
    for (int r = 0; r < 8; ++r)
      acc4[r & 3] += fast_exp2(fmaf(c[r], EXP2_SCALE, EXP2_BIAS));
  }
}

// ---------------------------------------------------------------------------
// Kernel 2: U = unmasked sum over tiles jt >= it. Block p handles row-tiles
// p and 511-p -> (512-p)+(p+1)=513 tiles/block, perfectly balanced.
// Contributes 2*U to the accumulator.
// ---------------------------------------------------------------------------
__global__ __launch_bounds__(256) void cl_gram_lse_kernel(
    const _Float16* __restrict__ xn, float* __restrict__ acc) {
  const int lane = threadIdx.x & 31;
  const int wave = threadIdx.x >> 5;
  const int sub  = lane & 15;
  const int hi   = lane >> 4;
  const int p    = blockIdx.x;            // 0..255

  float acc4[4] = {0.0f, 0.0f, 0.0f, 0.0f};
  cl_sweep_rowtile(xn, p,               wave, sub, hi, acc4);
  cl_sweep_rowtile(xn, N_TILES - 1 - p, wave, sub, hi, acc4);

  float local = ((acc4[0] + acc4[1]) + (acc4[2] + acc4[3])) * 2.0f;
#pragma unroll
  for (int off = 16; off > 0; off >>= 1) local += __shfl_xor(local, off, 32);
  if (lane == 0) atomicAdd(acc, local);
}

// ---------------------------------------------------------------------------
// Kernel 2b: exact diagonal correction, contributes -(T_d + D):
// each of the 512 diagonal tiles recomputed with the same WMMA path; each
// element subtracted once, true-diagonal elements subtracted twice.
// Total accumulated = 2U - T_d - D = sum over all off-diagonal entries.
// ---------------------------------------------------------------------------
__global__ __launch_bounds__(256) void cl_diagtile_kernel(
    const _Float16* __restrict__ xn, float* __restrict__ acc) {
  const int lane = threadIdx.x & 31;
  const int wave = threadIdx.x >> 5;
  const int sub  = lane & 15;
  const int hi   = lane >> 4;
  const int it   = blockIdx.x * 8 + wave;   // 64 blocks * 8 waves = 512 tiles

  const _Float16* row = xn + (size_t)(it * 16 + sub) * NF;
  v8f c = {};
#pragma unroll
  for (int kb = 0; kb < 4; ++kb) {
    v16h a = load_afrag(row, kb, hi);
    v16h b = *(const v16h*)(row + kb * 32 + hi * 16);
    c = __builtin_amdgcn_wmma_f32_16x16x32_f16(
            false, a, false, b, (short)0, c, false, false);
  }

  float local = 0.0f;
#pragma unroll
  for (int r = 0; r < 8; ++r) {
    const float e = fast_exp2(fmaf(c[r], EXP2_SCALE, EXP2_BIAS));
    local -= ((hi * 8 + r) == sub) ? 2.0f * e : e;   // -T_d term and -D term
  }
#pragma unroll
  for (int off = 16; off > 0; off >>= 1) local += __shfl_xor(local, off, 32);
  if (lane == 0) atomicAdd(acc, local);
}

// ---------------------------------------------------------------------------
// Kernel 3: loss = NEGATIVE_WEIGHT * (shift + log(sum))
// ---------------------------------------------------------------------------
__global__ void cl_finalize_kernel(const float* __restrict__ acc, float* __restrict__ out) {
  out[0] = INV_T + logf(acc[0]);
}

extern "C" void kernel_launch(void* const* d_in, const int* in_sizes, int n_in,
                              void* d_out, int out_size, void* d_ws, size_t ws_size,
                              hipStream_t stream) {
  (void)in_sizes; (void)n_in; (void)out_size; (void)ws_size;
  const float* x = (const float*)d_in[0];
  float* out = (float*)d_out;

  _Float16* xn = (_Float16*)d_ws;                                  // 2 MB
  float* acc = (float*)((char*)d_ws + (size_t)N_ROWS * NF * sizeof(_Float16));

  cl_normalize_kernel<<<N_ROWS / 8, 256, 0, stream>>>(x, xn, acc);
  cl_gram_lse_kernel<<<N_TILES / 2, 256, 0, stream>>>(xn, acc);
  cl_diagtile_kernel<<<N_TILES / 8, 256, 0, stream>>>(xn, acc);
  cl_finalize_kernel<<<1, 1, 0, stream>>>(acc, out);
}
